// PseudoTokenTEGridEncoder_86251533238883
// MI455X (gfx1250) — compile-verified, matrix operand order checked by reading, not run on hardware
//
#include <hip/hip_runtime.h>

// ---------------- constants (match reference shapes) ----------------
#define Bb    8
#define Nn    8192
#define Ec    128
#define Hc    8
#define HDc   16
#define KHc   32
#define G1c   32
#define G2c   32
#define Pc    32
#define Mc    1024           // G1*G2
#define Tc    8192           // Bb*Mc
#define NPTS  65536          // Bb*Nn

typedef __attribute__((ext_vector_type(16))) __bf16 v16bf;
typedef __attribute__((ext_vector_type(8)))  float  v8f;
union __align__(16) Frag { v16bf v; unsigned short s[16]; uint4 q[2]; };

// Weight swizzle: element (k,n) of a [128 x N] matrix -> fragment-ordered index.
// Lane (half,l16) of the wave owning n-tile nt reads 16 contiguous ushorts.
__device__ __forceinline__ int swzB(int k, int n) {
  return (((n >> 4) * 4 + (k >> 5)) * 512) + (((k >> 4) & 1) * 256) +
         ((n & 15) * 16) + (k & 15);
}

// ---------------- helpers ----------------
__device__ __forceinline__ unsigned encf(float f) {
  unsigned u = __float_as_uint(f);
  return (u & 0x80000000u) ? ~u : (u | 0x80000000u);
}
__device__ __forceinline__ float decf(unsigned m) {
  return __uint_as_float((m & 0x80000000u) ? (m & 0x7fffffffu) : ~m);
}
__device__ __forceinline__ unsigned short f2b(float f) {   // f32 -> bf16 RNE
  unsigned u = __float_as_uint(f);
  unsigned r = u + 0x7fffu + ((u >> 16) & 1u);
  return (unsigned short)(r >> 16);
}
__device__ __forceinline__ float b2f(unsigned short s) {
  return __uint_as_float(((unsigned)s) << 16);
}
__device__ __forceinline__ float wsum(float v) {
  for (int o = 16; o > 0; o >>= 1) v += __shfl_xor(v, o, 32);
  return v;
}
__device__ __forceinline__ float wmaxr(float v) {
  for (int o = 16; o > 0; o >>= 1) v = fmaxf(v, __shfl_xor(v, o, 32));
  return v;
}
__device__ __forceinline__ float wminr(float v) {
  for (int o = 16; o > 0; o >>= 1) v = fminf(v, __shfl_xor(v, o, 32));
  return v;
}

// ---------------- K0: init counters + minmax slots ----------------
__global__ void k_init(unsigned* mm, int* cnt) {
  int i = blockIdx.x * blockDim.x + threadIdx.x;
  if (i < 4) mm[i] = (i < 2) ? 0xFFFFFFFFu : 0u;
  for (int c = i; c < Tc; c += gridDim.x * blockDim.x) cnt[c] = 0;
}

// ---------------- K1: global per-dim min/max of x ----------------
__global__ void k_minmax(const float* __restrict__ x, unsigned* mm) {
  float mn0 = 3.4e38f, mn1 = 3.4e38f, mx0 = -3.4e38f, mx1 = -3.4e38f;
  for (int i = blockIdx.x * blockDim.x + threadIdx.x; i < NPTS;
       i += gridDim.x * blockDim.x) {
    float a = x[2 * i], b = x[2 * i + 1];
    mn0 = fminf(mn0, a); mx0 = fmaxf(mx0, a);
    mn1 = fminf(mn1, b); mx1 = fmaxf(mx1, b);
  }
  mn0 = wminr(mn0); mn1 = wminr(mn1); mx0 = wmaxr(mx0); mx1 = wmaxr(mx1);
  if ((threadIdx.x & 31) == 0) {
    atomicMin(&mm[0], encf(mn0)); atomicMin(&mm[1], encf(mn1));
    atomicMax(&mm[2], encf(mx0)); atomicMax(&mm[3], encf(mx1));
  }
}

// ---------------- K2: one-time small precompute (1 block x 128) ----------------
// q = LN_q(latent)@Wq ; k0/v0 = LN_k(latent)@{Wk,Wv} ; l0[h] = q.k0/4 + kb(0)
// Wcat[128x144] = [ Wv | Wk_h . q_h (8 cols) | zeros(8) ] (bf16, frag-swizzled)
// Wo/fW1/fW2 -> bf16 frag-swizzled ; bias_z1 = latent + bo ; v0 -> f32 ws
__global__ void k_prep(const float* latent, const float* Wq, const float* Wk,
                       const float* Wv, const float* Wo, const float* bo,
                       const float* kb1, const float* kW2, const float* kb2,
                       const float* lnqg, const float* lnqb,
                       const float* lnkg, const float* lnkb,
                       const float* fW1, const float* fW2,
                       float* v0f, float* l0ws, float* bias_z1,
                       unsigned short* Wcat, unsigned short* WoB,
                       unsigned short* fW1B, unsigned short* fW2B) {
  __shared__ float red[128], zqn[128], zkn[128], qv[128], k0v[128];
  int tid = threadIdx.x;
  float lv = latent[tid];
  red[tid] = lv; __syncthreads();
  for (int o = 64; o > 0; o >>= 1) { if (tid < o) red[tid] += red[tid + o]; __syncthreads(); }
  float mu = red[0] * (1.f / 128.f); __syncthreads();
  float d = lv - mu;
  red[tid] = d * d; __syncthreads();
  for (int o = 64; o > 0; o >>= 1) { if (tid < o) red[tid] += red[tid + o]; __syncthreads(); }
  float inv = rsqrtf(red[0] * (1.f / 128.f) + 1e-5f); __syncthreads();
  zqn[tid] = d * inv * lnqg[tid] + lnqb[tid];
  zkn[tid] = d * inv * lnkg[tid] + lnkb[tid];
  __syncthreads();
  float qa = 0.f, ka = 0.f, va = 0.f;
  for (int e = 0; e < 128; ++e) {
    float zq = zqn[e], zk = zkn[e];
    qa += zq * Wq[e * 128 + tid];
    ka += zk * Wk[e * 128 + tid];
    va += zk * Wv[e * 128 + tid];
  }
  qv[tid] = qa; k0v[tid] = ka; v0f[tid] = va;
  bias_z1[tid] = latent[tid] + bo[tid];
  __syncthreads();
  if (tid < Hc) {
    int h = tid; float dd = 0.f;
    for (int k = 0; k < HDc; ++k) dd += qv[h * HDc + k] * k0v[h * HDc + k];
    dd *= 0.25f;
    float kb0 = kb2[h];
    for (int u = 0; u < KHc; ++u) kb0 += fmaxf(kb1[u], 0.f) * kW2[u * Hc + h];
    l0ws[h] = dd + kb0;
  }
  // Wcat row k = tid (frag-swizzled writes)
  for (int c = 0; c < 128; ++c) Wcat[swzB(tid, c)] = f2b(Wv[tid * 128 + c]);
  for (int h = 0; h < Hc; ++h) {
    float wk = 0.f;
    for (int k = 0; k < HDc; ++k) wk += Wk[tid * 128 + h * HDc + k] * qv[h * HDc + k];
    Wcat[swzB(tid, 128 + h)] = f2b(wk);
  }
  for (int h = Hc; h < 16; ++h) Wcat[swzB(tid, 128 + h)] = 0;
  for (int i = tid; i < 128 * 128; i += 128) {
    int k = i >> 7, n = i & 127;
    int dpos = swzB(k, n);
    WoB[dpos] = f2b(Wo[i]); fW1B[dpos] = f2b(fW1[i]); fW2B[dpos] = f2b(fW2[i]);
  }
}

// ---------------- K3: point -> cell + slot (scatter indices) ----------------
__global__ void k_assign(const float* __restrict__ x, const unsigned* mm,
                         int* cnt, int* pts) {
  int i = blockIdx.x * blockDim.x + threadIdx.x;
  if (i >= NPTS) return;
  float mn0 = decf(mm[0]), mn1 = decf(mm[1]), mx0 = decf(mm[2]), mx1 = decf(mm[3]);
  float s0 = (mx0 - mn0) * (1.f / (G1c - 1)), s1 = (mx1 - mn1) * (1.f / (G2c - 1));
  float a = x[2 * i], b = x[2 * i + 1];
  int i0 = (int)fminf(fmaxf(rintf((a - mn0) / s0), 0.f), (float)(G1c - 1));
  int i1 = (int)fminf(fmaxf(rintf((b - mn1) / s1), 0.f), (float)(G2c - 1));
  int t = (i / Nn) * Mc + i0 * G2c + i1;
  int s = atomicAdd(&cnt[t], 1);
  if (s < Pc) pts[t * Pc + s] = i;
}

// ---------------- K4: row LayerNorm -> bf16 (one wave per 128-row) ----------------
__global__ void k_ln(const float* __restrict__ in, const float* g,
                     const float* bta, unsigned short* out, int M) {
  int wid = threadIdx.x >> 5, lane = threadIdx.x & 31;
  int row = blockIdx.x * (blockDim.x >> 5) + wid;
  if (row >= M) return;
  const float* r = in + (size_t)row * 128;
  float v[4]; float s = 0.f;
  for (int j = 0; j < 4; ++j) { v[j] = r[lane + 32 * j]; s += v[j]; }
  s = wsum(s);
  float mu = s * (1.f / 128.f);
  float q = 0.f;
  for (int j = 0; j < 4; ++j) { float d = v[j] - mu; q += d * d; }
  q = wsum(q);
  float inv = rsqrtf(q * (1.f / 128.f) + 1e-5f);
  for (int j = 0; j < 4; ++j) {
    int c = lane + 32 * j;
    out[(size_t)row * 128 + c] = f2b((v[j] - mu) * inv * g[c] + bta[c]);
  }
}

// ---------------- K5: big WMMA GEMM [M,128]x[128,144] -> V(bf16) + dots(f32) ----------------
__global__ __launch_bounds__(288) void k_gemm_cat(
    const unsigned short* __restrict__ A, const unsigned short* __restrict__ Bw,
    unsigned short* __restrict__ Vout, float* __restrict__ dots) {
  __shared__ __align__(16) unsigned short As[64 * 128];
  __shared__ __align__(16) unsigned short Bs[9 * 4 * 512];   // frag-ordered
  int tid = threadIdx.x;
  const size_t r0 = (size_t)blockIdx.x * 64;
  if (blockIdx.x + 1 < gridDim.x)
    __builtin_prefetch(A + (r0 + 64) * 128 + tid * 16, 0, 1);
  {
    const uint4* Ag = (const uint4*)(A + r0 * 128);
    const uint4* Bg = (const uint4*)Bw;
    uint4* Al = (uint4*)As;
    uint4* Bl = (uint4*)Bs;
    for (int i = tid; i < 64 * 128 / 8; i += 288) Al[i] = Ag[i];
    for (int i = tid; i < 9 * 4 * 512 / 8; i += 288) Bl[i] = Bg[i];
  }
  __syncthreads();
  int w = tid >> 5, lane = tid & 31, half = lane >> 4, l16 = lane & 15;
  int n0 = w * 16;
  Frag bf[4];
  for (int ks = 0; ks < 4; ++ks) {
    const uint4* p = (const uint4*)&Bs[(w * 4 + ks) * 512 + lane * 16];
    bf[ks].q[0] = p[0]; bf[ks].q[1] = p[1];
  }
  for (int mt = 0; mt < 4; ++mt) {
    Frag af[4];
    for (int ks = 0; ks < 4; ++ks) {
      const uint4* p = (const uint4*)&As[(mt * 16 + l16) * 128 + ks * 32 + half * 16];
      af[ks].q[0] = p[0]; af[ks].q[1] = p[1];
    }
    v8f acc; for (int i = 0; i < 8; ++i) acc[i] = 0.f;
    for (int ks = 0; ks < 4; ++ks)
      acc = __builtin_amdgcn_wmma_f32_16x16x32_bf16(
          false, af[ks].v, false, bf[ks].v, (short)0, acc, false, false);
    if (w < 8) {              // wave-uniform branch: no EXEC churn
      for (int r = 0; r < 8; ++r) {
        size_t row = r0 + mt * 16 + r + 8 * half;
        Vout[row * 128 + n0 + l16] = f2b(acc[r]);
      }
    } else if (l16 < Hc) {
      for (int r = 0; r < 8; ++r) {
        size_t row = r0 + mt * 16 + r + 8 * half;
        dots[row * 8 + l16] = 0.25f * acc[r];
      }
    }
  }
}

// ---------------- K7: generic WMMA GEMM [M,128]x[128,128] + bias/resid/relu ----
template <int RELU, int HAS_RESID, int HAS_OUTF, int HAS_OUTB>
__global__ __launch_bounds__(256) void k_gemm128(
    const unsigned short* __restrict__ A, const unsigned short* __restrict__ Bw,
    const float* __restrict__ bias, const float* __restrict__ resid,
    float* __restrict__ outF, unsigned short* __restrict__ outB) {
  __shared__ __align__(16) unsigned short As[64 * 128];
  __shared__ __align__(16) unsigned short Bs[8 * 4 * 512];   // frag-ordered
  int tid = threadIdx.x;
  const size_t r0 = (size_t)blockIdx.x * 64;
  if (blockIdx.x + 1 < gridDim.x)
    __builtin_prefetch(A + (r0 + 64) * 128 + tid * 32, 0, 1);
  {
    const uint4* Ag = (const uint4*)(A + r0 * 128);
    const uint4* Bg = (const uint4*)Bw;
    uint4* Al = (uint4*)As;
    uint4* Bl = (uint4*)Bs;
    for (int i = tid; i < 64 * 128 / 8; i += 256) Al[i] = Ag[i];
    for (int i = tid; i < 8 * 4 * 512 / 8; i += 256) Bl[i] = Bg[i];
  }
  __syncthreads();
  int w = tid >> 5, lane = tid & 31, half = lane >> 4, l16 = lane & 15;
  int n0 = w * 16;
  float bcol = bias[n0 + l16];
  Frag bf[4];
  for (int ks = 0; ks < 4; ++ks) {
    const uint4* p = (const uint4*)&Bs[(w * 4 + ks) * 512 + lane * 16];
    bf[ks].q[0] = p[0]; bf[ks].q[1] = p[1];
  }
  for (int mt = 0; mt < 4; ++mt) {
    Frag af[4];
    for (int ks = 0; ks < 4; ++ks) {
      const uint4* p = (const uint4*)&As[(mt * 16 + l16) * 128 + ks * 32 + half * 16];
      af[ks].q[0] = p[0]; af[ks].q[1] = p[1];
    }
    v8f acc; for (int i = 0; i < 8; ++i) acc[i] = 0.f;
    for (int ks = 0; ks < 4; ++ks)
      acc = __builtin_amdgcn_wmma_f32_16x16x32_bf16(
          false, af[ks].v, false, bf[ks].v, (short)0, acc, false, false);
    for (int r = 0; r < 8; ++r) {
      size_t row = r0 + mt * 16 + r + 8 * half;
      int col = n0 + l16;
      float val = acc[r] + bcol;
      if (HAS_RESID) val += resid[row * 128 + col];
      if (RELU) val = fmaxf(val, 0.f);
      if (HAS_OUTF) outF[row * 128 + col] = val;
      if (HAS_OUTB) outB[row * 128 + col] = f2b(val);
    }
  }
}

// ---------------- K6: per-cell attention (one wave32 per cell) ----------------
__global__ __launch_bounds__(32) void k_attn(
    const float* __restrict__ x, const int* __restrict__ cnt,
    const int* __restrict__ pts, const float* __restrict__ dotsP,
    const unsigned short* __restrict__ Vb, const float* __restrict__ v0f,
    const float* __restrict__ l0, const unsigned* mm,
    const float* __restrict__ kW1, const float* __restrict__ kb1,
    const float* __restrict__ kW2, const float* __restrict__ kb2,
    unsigned short* __restrict__ OUTb, float* __restrict__ xout) {
  __shared__ float sA[Hc][Pc + 1];
  __shared__ int sPt[Pc];
  int t = blockIdx.x, lane = threadIdx.x;
  int c = cnt[t]; int count = c < Pc ? c : Pc;
  int m = t & (Mc - 1); int c0 = m >> 5, c1 = m & 31;
  float mn0 = decf(mm[0]), mn1 = decf(mm[1]), mx0 = decf(mm[2]), mx1 = decf(mm[3]);
  float s0 = (mx0 - mn0) * (1.f / (G1c - 1)), s1 = (mx1 - mn1) * (1.f / (G2c - 1));
  float gx = mn0 + c0 * s0, gy = mn1 + c1 * s1;
  int pt = -1; float px = 0.f, py = 0.f;
  if (lane < count) { pt = pts[t * Pc + lane]; px = x[2 * pt]; py = x[2 * pt + 1]; }
  sPt[lane] = pt;
  float kb[Hc];
  for (int h = 0; h < Hc; ++h) kb[h] = kb2[h];
  if (lane < count) {
    float d0 = px - gx, d1 = py - gy;
    for (int u = 0; u < KHc; ++u) {
      float hh = fmaxf(d0 * kW1[u] + d1 * kW1[KHc + u] + kb1[u], 0.f);
      for (int h = 0; h < Hc; ++h) kb[h] += hh * kW2[u * Hc + h];
    }
  }
  float mA = 0.f, mA0 = 0.f;
  for (int h = 0; h < Hc; ++h) {
    float l0h = l0[h];
    float lg = (lane < count) ? (dotsP[(size_t)pt * 8 + h] + kb[h]) : -1e30f;
    float mx = fmaxf(wmaxr(lg), l0h);
    float e = (lane < count) ? __expf(lg - mx) : 0.f;
    float e0 = __expf(l0h - mx);
    float inv = 1.f / (wsum(e) + e0);
    float a = e * inv, a0 = e0 * inv;
    sA[h][lane + 1] = a;
    if (lane == 0) sA[h][0] = a0;
    mA += a * 0.125f; mA0 += a0 * 0.125f;
  }
  float xs0 = wsum(mA * px) + mA0 * gx;
  float xs1 = wsum(mA * py) + mA0 * gy;
  if (lane == 0) { xout[2 * t] = xs0; xout[2 * t + 1] = xs1; }
  __syncthreads();
  for (int cc = 0; cc < 4; ++cc) {
    int col = lane + cc * 32; int h = col >> 4;
    float acc = sA[h][0] * v0f[col];
    for (int j = 0; j < count; ++j)
      acc += sA[h][j + 1] * b2f(Vb[(size_t)sPt[j] * 128 + col]);
    OUTb[(size_t)t * 128 + col] = f2b(acc);
  }
}

// ---------------- workspace layout (bytes, all 256-aligned) ----------------
#define OFF_MM    0u
#define OFF_CNT   256u
#define OFF_V0    33024u
#define OFF_L0    33536u
#define OFF_BZ1   33792u
#define OFF_WCAT  34304u
#define OFF_WO    71168u
#define OFF_FW1   103936u
#define OFF_FW2   136704u
#define OFF_ZN    169472u                 // 16 MB  (reused for T1/H later)
#define OFF_V     16946688u               // 16 MB
#define OFF_DOTS  33723904u               // 2 MB
#define OFF_PTS   35821056u               // 1 MB
#define OFF_OUT   36869632u               // 2 MB
#define OFF_Z1    38966784u               // 4 MB
#define OFF_T1    OFF_ZN                  // reuse zn region
#define OFF_H     (OFF_ZN + 2097152u)

extern "C" void kernel_launch(void* const* d_in, const int* in_sizes, int n_in,
                              void* d_out, int out_size, void* d_ws, size_t ws_size,
                              hipStream_t stream) {
  (void)in_sizes; (void)n_in; (void)out_size; (void)ws_size;
  const float* x      = (const float*)d_in[0];
  const float* z      = (const float*)d_in[1];
  const float* latent = (const float*)d_in[6];
  const float* Wq     = (const float*)d_in[7];
  const float* Wk     = (const float*)d_in[8];
  const float* Wv     = (const float*)d_in[9];
  const float* Wo     = (const float*)d_in[10];
  const float* bo     = (const float*)d_in[11];
  const float* kW1    = (const float*)d_in[12];
  const float* kb1    = (const float*)d_in[13];
  const float* kW2    = (const float*)d_in[14];
  const float* kb2    = (const float*)d_in[15];
  const float* lnqg   = (const float*)d_in[16];
  const float* lnqb   = (const float*)d_in[17];
  const float* lnkg   = (const float*)d_in[18];
  const float* lnkb   = (const float*)d_in[19];
  const float* ln2g   = (const float*)d_in[20];
  const float* ln2b   = (const float*)d_in[21];
  const float* fW1    = (const float*)d_in[22];
  const float* fb1    = (const float*)d_in[23];
  const float* fW2    = (const float*)d_in[24];
  const float* fb2    = (const float*)d_in[25];

  char* ws = (char*)d_ws;
  unsigned*       mm    = (unsigned*)(ws + OFF_MM);
  int*            cnt   = (int*)(ws + OFF_CNT);
  float*          v0f   = (float*)(ws + OFF_V0);
  float*          l0ws  = (float*)(ws + OFF_L0);
  float*          bz1   = (float*)(ws + OFF_BZ1);
  unsigned short* Wcat  = (unsigned short*)(ws + OFF_WCAT);
  unsigned short* WoB   = (unsigned short*)(ws + OFF_WO);
  unsigned short* fW1B  = (unsigned short*)(ws + OFF_FW1);
  unsigned short* fW2B  = (unsigned short*)(ws + OFF_FW2);
  unsigned short* znB   = (unsigned short*)(ws + OFF_ZN);
  unsigned short* Vb    = (unsigned short*)(ws + OFF_V);
  float*          dots  = (float*)(ws + OFF_DOTS);
  int*            pts   = (int*)(ws + OFF_PTS);
  unsigned short* OUTb  = (unsigned short*)(ws + OFF_OUT);
  float*          z1    = (float*)(ws + OFF_Z1);
  unsigned short* t1B   = (unsigned short*)(ws + OFF_T1);
  unsigned short* Hb    = (unsigned short*)(ws + OFF_H);

  float* xout = (float*)d_out;                       // [B,32,32,2]
  float* zout = (float*)d_out + (size_t)Tc * 2;      // [B,32,32,128]

  k_init<<<32, 256, 0, stream>>>(mm, cnt);
  k_minmax<<<256, 256, 0, stream>>>(x, mm);
  k_prep<<<1, 128, 0, stream>>>(latent, Wq, Wk, Wv, Wo, bo, kb1, kW2, kb2,
                                lnqg, lnqb, lnkg, lnkb, fW1, fW2,
                                v0f, l0ws, bz1, Wcat, WoB, fW1B, fW2B);
  k_assign<<<NPTS / 256, 256, 0, stream>>>(x, mm, cnt, pts);
  k_ln<<<NPTS / 8, 256, 0, stream>>>(z, lnkg, lnkb, znB, NPTS);
  k_gemm_cat<<<NPTS / 64, 288, 0, stream>>>(znB, Wcat, Vb, dots);
  k_attn<<<Tc, 32, 0, stream>>>(x, cnt, pts, dots, Vb, v0f, l0ws, mm,
                                kW1, kb1, kW2, kb2, OUTb, xout);
  // z1 = OUT @ Wo + (latent + bo)
  k_gemm128<0, 0, 1, 0><<<Tc / 64, 256, 0, stream>>>(OUTb, WoB, bz1, nullptr, z1, nullptr);
  // t1 = LN2(z1)
  k_ln<<<Tc / 8, 256, 0, stream>>>(z1, ln2g, ln2b, t1B, Tc);
  // H = relu(t1 @ fW1 + fb1)
  k_gemm128<1, 0, 0, 1><<<Tc / 64, 256, 0, stream>>>(t1B, fW1B, fb1, nullptr, nullptr, Hb);
  // z_out = z1 + H @ fW2 + fb2
  k_gemm128<0, 1, 1, 0><<<Tc / 64, 256, 0, stream>>>(Hb, fW2B, fb2, z1, zout, nullptr);
}
